// deepGoPlus_66537633349904
// MI455X (gfx1250) — matrix operand order repaired and checked by reading, not compiled, not run on hardware
//
#include <hip/hip_runtime.h>
#include <stdint.h>

typedef __attribute__((ext_vector_type(16))) _Float16 v16h;
typedef __attribute__((ext_vector_type(8)))  float    v8f;

#define NBR   37
#define CIN   4
#define LEN   200
#define COUT  512
#define BATCH 1024
#define FEATS (NBR * COUT)   // 18944
#define MAXNC 10             // ceil(78/8)

struct Ptrs37 { const float* p[NBR]; };
struct Ints37 { int v[NBR]; };

// single-instruction max (avoids fmaxf NaN-canonicalization self-maxes)
__device__ __forceinline__ float fmax32(float a, float b) {
    float r;
    asm("v_max_num_f32 %0, %1, %2" : "=v"(r) : "v"(a), "v"(b));
    return r;
}

// ---------------------------------------------------------------- x -> f16
__global__ __launch_bounds__(256) void k_cvt_x(const float* __restrict__ x,
                                               _Float16* __restrict__ xh, int n) {
    int i = blockIdx.x * blockDim.x + threadIdx.x;
    if (i < n) xh[i] = (_Float16)x[i];
}

// ------------------------------------------------- pack weights to A-frags
// frag g (= branch_base + ct*nc + c), lane l (0..31), elem e (0..15)
// value = W[br][co = ct*16 + (l&15)][kk = 32c + K(e, l>>4)], CDNA5 A-layout.
__global__ __launch_bounds__(256) void k_pack_w(Ptrs37 wsrc,
                                                _Float16* __restrict__ wp, int total) {
    int idx = blockIdx.x * blockDim.x + threadIdx.x;
    if (idx >= total) return;
    int e = idx & 15;
    int l = (idx >> 4) & 31;
    int g = idx >> 9;
    int br = 0, base = 0;
    for (; br < NBR; ++br) {
        int k   = 6 + 2 * br;
        int fpb = 32 * ((k + 7) >> 3);
        if (g < base + fpb) break;
        base += fpb;
    }
    int k  = 6 + 2 * br;
    int nc = (k + 7) >> 3;
    int local = g - base;
    int ct = local / nc;
    int c  = local - ct * nc;
    int h  = l >> 4;
    int co = ct * 16 + (l & 15);
    int K  = e + 8 * h + ((e >= 8) ? 8 : 0);    // 16-bit A-matrix VGPR layout
    int kk = 32 * c + K;
    float v = 0.0f;
    if (kk < 4 * k) {
        int ci = kk / k;
        int j  = kk - ci * k;
        v = wsrc.p[br][((size_t)co * 4 + ci) * k + j];
    }
    wp[idx] = (_Float16)v;
}

// -------------------------------------------- fused conv + bias + max (WMMA)
template <int K>
__device__ __forceinline__ void conv_branch(
    _Float16* __restrict__ smem,         // >= 800 + MAXNC*512 halfs
    const _Float16* __restrict__ xh,     // [B][4][200] f16
    const _Float16* __restrict__ wpb,    // this branch's packed A fragments
    const float*    __restrict__ bias,   // [512]
    _Float16*       __restrict__ houtb)  // h + br*512; batch stride FEATS
{
    constexpr int NC   = (K + 7) >> 3;   // K-chunks of 32
    constexpr int LOUT = 201 - K;
    constexpr int NT   = (LOUT + 15) >> 4;
    constexpr int KMAX = 4 * K;
    constexpr int NEL  = NC * 2;         // B-tile gather elems per thread

    _Float16* xs = smem;                 // 800 halfs: x for batch b
    _Float16* bt = smem + CIN * LEN;     // NC*512 halfs: B tile, WMMA reg order

    const int b   = blockIdx.x;
    const int tid = threadIdx.x;

    {   // stage x[b] (1600 B) into LDS, 4 halfs (8 B) per thread
        int i0 = tid * 4;
        if (i0 < CIN * LEN)
            *(uint64_t*)&xs[i0] = *(const uint64_t*)(xh + (size_t)b * (CIN * LEN) + i0);
    }

    // t-tile-invariant gather offsets for the B-tile build (registers)
    int srcoff[NEL], colv[NEL];
    #pragma unroll
    for (int m = 0; m < NEL; ++m) {
        int idx = tid + m * 256;
        int e   = idx & 15;
        int l   = (idx >> 4) & 31;
        int c   = idx >> 9;
        int kk  = 32 * c + 8 * (l >> 4) + e + ((e >= 8) ? 8 : 0); // B layout: K = 32c + 8h + e (+8 if e>=8)
        int kkc = (kk < KMAX) ? kk : (KMAX - 1);    // padded region: A is zero
        int ci  = kkc / K;
        srcoff[m] = ci * LEN + (kkc - ci * K);
        colv[m]   = l & 15;
    }

    const int lane = tid & 31;
    const int wave = tid >> 5;           // 0..7
    const int h    = lane >> 4;
    const int ln   = lane & 15;

    float mx[4][8];
    #pragma unroll
    for (int g = 0; g < 4; ++g)
        #pragma unroll
        for (int i = 0; i < 8; ++i) mx[g][i] = -3.4e38f;

    __syncthreads();

    #pragma unroll 1
    for (int tt = 0; tt < NT; ++tt) {
        // cooperative B-tile build (shared by all 32 co-tiles)
        #pragma unroll
        for (int m = 0; m < NEL; ++m) {
            int t = tt * 16 + colv[m];
            if (t > LOUT - 1) t = LOUT - 1;  // duplicate column: safe under max
            bt[tid + m * 256] = xs[srcoff[m] + t];
        }
        __syncthreads();
        #pragma unroll
        for (int g = 0; g < 4; ++g) {
            const int ct = wave + g * 8;     // co tile
            v8f acc = {};
            #pragma unroll
            for (int c = 0; c < NC; ++c) {
                const v16h a  = *(const v16h*)(wpb + (((ct * NC + c) * 32 + lane) << 4));
                const v16h bb = *(const v16h*)(bt  + (((c * 32) + lane) << 4));
                acc = __builtin_amdgcn_wmma_f32_16x16x32_f16(
                        false, a, false, bb, (short)0, acc, false, false);
            }
            #pragma unroll
            for (int i = 0; i < 8; ++i) mx[g][i] = fmax32(mx[g][i], acc[i]);
        }
        __syncthreads();
    }

    // max over t: reduce across the 16 lanes of each half-wave, add bias, store
    #pragma unroll
    for (int g = 0; g < 4; ++g) {
        const int ct = wave + g * 8;
        #pragma unroll
        for (int i = 0; i < 8; ++i) {
            float v = mx[g][i];
            v = fmax32(v, __shfl_xor(v, 1, 32));
            v = fmax32(v, __shfl_xor(v, 2, 32));
            v = fmax32(v, __shfl_xor(v, 4, 32));
            v = fmax32(v, __shfl_xor(v, 8, 32));
            if (ln == 0) {                    // lanes 0 and 16 each cover 8 co
                int co = ct * 16 + i + 8 * h; // C/D layout: m = reg + 8*half
                houtb[(size_t)b * FEATS + co] = (_Float16)(v + bias[co]);
            }
        }
    }
}

// one launch for all 37 branches: blockIdx.y selects the specialization
__global__ __launch_bounds__(256) void k_conv_all(
    const _Float16* __restrict__ xh, const _Float16* __restrict__ wp,
    Ptrs37 bias, Ints37 fb, _Float16* __restrict__ hb)
{
    __shared__ __align__(16) _Float16 smem[CIN * LEN + MAXNC * 512];
    const int br = blockIdx.y;
    switch (br) {
    #define BR_CASE(i)                                                          \
        case (i):                                                               \
            conv_branch<6 + 2 * (i)>(smem, xh, wp + (size_t)fb.v[(i)] * 512,    \
                                     bias.p[(i)], hb + (size_t)(i) * COUT);     \
            break;
    BR_CASE(0)  BR_CASE(1)  BR_CASE(2)  BR_CASE(3)  BR_CASE(4)  BR_CASE(5)
    BR_CASE(6)  BR_CASE(7)  BR_CASE(8)  BR_CASE(9)  BR_CASE(10) BR_CASE(11)
    BR_CASE(12) BR_CASE(13) BR_CASE(14) BR_CASE(15) BR_CASE(16) BR_CASE(17)
    BR_CASE(18) BR_CASE(19) BR_CASE(20) BR_CASE(21) BR_CASE(22) BR_CASE(23)
    BR_CASE(24) BR_CASE(25) BR_CASE(26) BR_CASE(27) BR_CASE(28) BR_CASE(29)
    BR_CASE(30) BR_CASE(31) BR_CASE(32) BR_CASE(33) BR_CASE(34) BR_CASE(35)
    BR_CASE(36)
    #undef BR_CASE
    default: break;
    }
}

// ------------------------------------------------------------ final fc dot
__global__ __launch_bounds__(256) void k_fc(const _Float16* __restrict__ hin,
                                            const float* __restrict__ fcw,
                                            const float* __restrict__ fcb,
                                            float* __restrict__ out) {
    __shared__ float red[256];
    const int b   = blockIdx.x;
    const int tid = threadIdx.x;
    const _Float16* hb = hin + (size_t)b * FEATS;
    float s = 0.0f;
    for (int f = tid; f < FEATS; f += 256) s += (float)hb[f] * fcw[f];
    red[tid] = s;
    __syncthreads();
    for (int off = 128; off > 0; off >>= 1) {
        if (tid < off) red[tid] += red[tid + off];
        __syncthreads();
    }
    if (tid == 0) out[b] = red[0] + fcb[0];
}

// ---------------------------------------------------------------- launcher
extern "C" void kernel_launch(void* const* d_in, const int* in_sizes, int n_in,
                              void* d_out, int out_size, void* d_ws, size_t ws_size,
                              hipStream_t stream) {
    (void)in_sizes; (void)n_in; (void)out_size; (void)ws_size;
    // dict order: x, conv_ws[37], conv_bs[37], fc_w, fc_b
    const float* x   = (const float*)d_in[0];
    Ptrs37 wptrs, bptrs;
    for (int i = 0; i < NBR; ++i) {
        wptrs.p[i] = (const float*)d_in[1 + i];
        bptrs.p[i] = (const float*)d_in[1 + NBR + i];
    }
    const float* fcw = (const float*)d_in[1 + 2 * NBR];
    const float* fcb = (const float*)d_in[2 + 2 * NBR];

    // per-branch fragment bases: fbase[i] = 32 * sum_{j<i} ceil(k_j/8)
    Ints37 fb;
    int facc = 0;
    for (int i = 0; i < NBR; ++i) {
        fb.v[i] = facc;
        facc += 32 * (((6 + 2 * i) + 7) >> 3);
    }   // facc == 6656 fragments total

    const int    wp_elems = facc * 512;                        // 3,407,872 halfs
    const size_t xh_bytes = (size_t)BATCH * CIN * LEN * 2;     // 1,638,400
    const size_t wp_bytes = (size_t)wp_elems * 2;              // 6,815,744

    char* ws = (char*)d_ws;
    _Float16* xh = (_Float16*)ws;
    _Float16* wp = (_Float16*)(ws + xh_bytes);
    _Float16* hb = (_Float16*)(ws + xh_bytes + wp_bytes);      // [B][37][512] f16

    const int nx = BATCH * CIN * LEN;
    k_cvt_x<<<(nx + 255) / 256, 256, 0, stream>>>(x, xh, nx);
    k_pack_w<<<(wp_elems + 255) / 256, 256, 0, stream>>>(wptrs, wp, wp_elems);
    dim3 grid(BATCH, NBR);
    k_conv_all<<<grid, 256, 0, stream>>>(xh, wp, bptrs, fb, hb);
    k_fc<<<BATCH, 256, 0, stream>>>(hb, fcw, fcb, (float*)d_out);
}
